// HMM_DPM_1924145348738
// MI455X (gfx1250) — matrix operand order repaired and checked by reading, not compiled, not run on hardware
//
#include <hip/hip_runtime.h>
#include <hip/hip_bf16.h>
#include <math.h>

typedef __attribute__((ext_vector_type(16))) _Float16 v16h;
typedef __attribute__((ext_vector_type(8)))  _Float16 v8h;
typedef __attribute__((ext_vector_type(8)))  float    v8f;

constexpr int H      = 256;
constexpr int NSTEPS = 100;
constexpr int Bb = 8, Ll = 1024;
constexpr int Nn  = Bb * Ll;          // 8192
constexpr int Kk  = 4, Ss = 3;
constexpr int KSN = Kk * Ss * Nn;     // 98304

// ---- workspace layout (float offsets into d_ws) ----
constexpr int SQA  = 0;       // sqrt(alpha_bar)[100] (padded 128)
constexpr int SQB  = 128;     // sqrt(1-alpha_bar)[100]
constexpr int BNS  = 256;     // bn_g / sqrt(1+eps)  [512]
constexpr int C0O  = 768;     // p_b @ m_w1[0:256] + m_b1  [512]
constexpr int KTO  = 1280;    // emb @ m_w1[512:768]       [4*512]
constexpr int TTO  = 3328;    // time-MLP @ m_w1[256:512]  [100*512]
constexpr int MSEO = 54528;   // per-row mse               [98304]
// packed f16 weights (B-fragment-major), byte offsets
constexpr size_t W1P_B = 611328;          // 64x512  halves (fused p_w@m_w1a)
constexpr size_t W2P_B = W1P_B + 65536;   // 512x256 halves
constexpr size_t W3P_B = W2P_B + 262144;  // 256x64  halves

// silu with hardware reciprocal: x * v_rcp(1 + exp(-x))
__device__ __forceinline__ float silu_f(float v) {
  return v * __builtin_amdgcn_rcpf(1.f + __expf(-v));
}

// ---------------- setup kernels ----------------

__global__ void k_sched(const float* __restrict__ bn_g, float* __restrict__ ws) {
  int tid = threadIdx.x;
  if (tid == 0) {
    float prod = 1.f;
    for (int i = 0; i < NSTEPS; ++i) {
      float beta = 1e-4f + (0.02f - 1e-4f) * (float)i / 99.f;
      prod *= (1.f - beta);
      ws[SQA + i] = sqrtf(prod);
      ws[SQB + i] = sqrtf(1.f - prod);
    }
  }
  for (int c = tid; c < 512; c += blockDim.x)
    ws[BNS + c] = bn_g[c] * rsqrtf(1.f + 1e-5f);
}

// block 0: c0 = p_b @ m_w1[0:256,:] + m_b1 ; blocks 1..4: k_table
__global__ void k_c0kt(const float* __restrict__ p_b, const float* __restrict__ m_b1,
                       const float* __restrict__ emb, const float* __restrict__ m_w1,
                       float* __restrict__ ws) {
  int c = threadIdx.x;            // 0..511
  int bid = blockIdx.x;
  if (bid == 0) {
    float acc = m_b1[c];
    for (int i = 0; i < 256; ++i) acc += p_b[i] * m_w1[i * 512 + c];
    ws[C0O + c] = acc;
  } else {
    int k = bid - 1;
    float acc = 0.f;
    for (int i = 0; i < 256; ++i) acc += emb[k * 256 + i] * m_w1[(512 + i) * 512 + c];
    ws[KTO + k * 512 + c] = acc;
  }
}

// one block per t value: sinusoidal -> time MLP -> fold through m_w1[256:512,:]
__global__ void k_ttable(const float* __restrict__ t_w1, const float* __restrict__ t_b1,
                         const float* __restrict__ t_w2, const float* __restrict__ t_b2,
                         const float* __restrict__ m_w1, float* __restrict__ ws) {
  __shared__ float sb[256], l1[256], l2[256];
  int tid = threadIdx.x;          // 0..511
  float tf = (float)blockIdx.x;
  if (tid < 128) {
    float fr = __expf((float)tid * (-logf(10000.f) / 127.f));
    float ang = tf * fr;
    sb[tid]       = sinf(ang);
    sb[tid + 128] = cosf(ang);
  }
  __syncthreads();
  if (tid < 256) {
    float acc = t_b1[tid];
    for (int i = 0; i < 256; ++i) acc += sb[i] * t_w1[i * 256 + tid];
    l1[tid] = silu_f(acc);
  }
  __syncthreads();
  if (tid < 256) {
    float acc = t_b2[tid];
    for (int i = 0; i < 256; ++i) acc += l1[i] * t_w2[i * 256 + tid];
    l2[tid] = acc;
  }
  __syncthreads();
  float acc = 0.f;
  for (int i = 0; i < 256; ++i) acc += l2[i] * m_w1[(256 + i) * 512 + tid];
  ws[TTO + blockIdx.x * 512 + tid] = acc;
}

// B-fragment packing: element (k,col) -> [(kstep*NT+ntile)*32 + lane]*16 + j
// with kstep=k>>5, g=(k>>3)&1, j=((k>>4)&1)*8+(k&7), ntile=col>>4, lane=g*16+(col&15)
__device__ __forceinline__ int bpack_idx(int k, int col, int NT) {
  int kstep = k >> 5, g = (k >> 3) & 1, j = ((k >> 4) & 1) * 8 + (k & 7);
  int nt = col >> 4, lane = g * 16 + (col & 15);
  return (((kstep * NT + nt) * 32) + lane) * 16 + j;
}

// fused W1 = p_w[64,256] @ m_w1[0:256, 0:512], packed f16
__global__ void k_w1pack(const float* __restrict__ p_w, const float* __restrict__ m_w1,
                         _Float16* __restrict__ w1p) {
  int d = blockIdx.x;             // 0..63  (K dim of GEMM1)
  int c = threadIdx.x;            // 0..511
  float acc = 0.f;
  for (int i = 0; i < 256; ++i) acc += p_w[d * 256 + i] * m_w1[i * 512 + c];
  w1p[bpack_idx(d, c, 32)] = (_Float16)acc;
}

__global__ void k_w2pack(const float* __restrict__ m_w2, _Float16* __restrict__ w2p) {
  int k = blockIdx.x;             // 0..511
  int c = threadIdx.x;            // 0..255
  w2p[bpack_idx(k, c, 16)] = (_Float16)m_w2[k * 256 + c];
}

__global__ void k_w3pack(const float* __restrict__ m_w3, _Float16* __restrict__ w3p) {
  int k = blockIdx.x;             // 0..255
  int c = threadIdx.x;            // 0..63
  w3p[bpack_idx(k, c, 4)] = (_Float16)m_w3[k * 64 + c];
}

// ---------------- fused emission kernel ----------------
// 64 threads = 2 waves; each wave owns a 16-row tile of the [KSN, .] batch.
// Both waves stream identical weight-fragment sequences -> WGP$ captures reuse.
__global__ void __launch_bounds__(64) k_main(
    const float* __restrict__ x, const int* __restrict__ t_idx,
    const float* __restrict__ noise, const float* __restrict__ bn_b,
    const float* __restrict__ m_b2, const float* __restrict__ m_b3,
    const float* __restrict__ ws,
    const _Float16* __restrict__ w1p, const _Float16* __restrict__ w2p,
    const _Float16* __restrict__ w3p, float* __restrict__ mse_out) {
  __shared__ __align__(16) _Float16 h1s[2][16][512];   // 32 KB
  __shared__ __align__(16) _Float16 h2s[2][16][256];   // 16 KB
  __shared__ int meta_t[2][16];
  __shared__ int meta_k[2][16];

  const int tid  = threadIdx.x;
  const int w    = tid >> 5;
  const int lane = tid & 31;
  const int m    = lane & 15;     // A-row / B-col within tile
  const int g    = lane >> 4;     // lane half
  const int rowbase = blockIdx.x * 32 + w * 16;
  const int r   = rowbase + m;                 // flattened [K,S,N] row
  const int n   = r & (Nn - 1);
  const int kst = r / (Ss * Nn);
  const int t   = t_idx[r];
  if (g == 0) { meta_t[w][m] = t; meta_k[w][m] = kst; }
  __syncthreads();

  // ---- build A fragments of x_t (16 rows x 64 K, f16) ----
  const float sa = ws[SQA + t], sb2 = ws[SQB + t];
  const float* xr = x + n * 64;
  const float* nr = noise + r * 64;
  v16h A0, A1;
#pragma unroll
  for (int f = 0; f < 2; ++f) {
    v16h a;
#pragma unroll
    for (int c2 = 0; c2 < 2; ++c2) {
      int d0 = f * 32 + c2 * 16 + g * 8;       // K index of half j = c2*8
      float4 xv0 = *(const float4*)(xr + d0);
      float4 xv1 = *(const float4*)(xr + d0 + 4);
      float4 nv0 = *(const float4*)(nr + d0);
      float4 nv1 = *(const float4*)(nr + d0 + 4);
      a[c2 * 8 + 0] = (_Float16)(sa * xv0.x + sb2 * nv0.x);
      a[c2 * 8 + 1] = (_Float16)(sa * xv0.y + sb2 * nv0.y);
      a[c2 * 8 + 2] = (_Float16)(sa * xv0.z + sb2 * nv0.z);
      a[c2 * 8 + 3] = (_Float16)(sa * xv0.w + sb2 * nv0.w);
      a[c2 * 8 + 4] = (_Float16)(sa * xv1.x + sb2 * nv1.x);
      a[c2 * 8 + 5] = (_Float16)(sa * xv1.y + sb2 * nv1.y);
      a[c2 * 8 + 6] = (_Float16)(sa * xv1.z + sb2 * nv1.z);
      a[c2 * 8 + 7] = (_Float16)(sa * xv1.w + sb2 * nv1.w);
    }
    if (f == 0) A0 = a; else A1 = a;
  }

  // ---- GEMM1: h1 = bn(silu(x_t @ Wfused + c0 + t_table[t] + k_table[k])) ----
  for (int nt = 0; nt < 32; ++nt) {
    v8f acc = {};
    v16h b0 = *(const v16h*)(w1p + ((0 * 32 + nt) * 32 + lane) * 16);
    acc = __builtin_amdgcn_wmma_f32_16x16x32_f16(false, A0, false, b0, (short)0, acc, false, false);
    v16h b1 = *(const v16h*)(w1p + ((1 * 32 + nt) * 32 + lane) * 16);
    acc = __builtin_amdgcn_wmma_f32_16x16x32_f16(false, A1, false, b1, (short)0, acc, false, false);
    int ncol = nt * 16 + m;
    float c0v = ws[C0O + ncol], bns = ws[BNS + ncol], bnbv = bn_b[ncol];
#pragma unroll
    for (int v = 0; v < 8; ++v) {
      int row = v + 8 * g;
      float val = acc[v] + c0v + ws[TTO + meta_t[w][row] * 512 + ncol]
                               + ws[KTO + meta_k[w][row] * 512 + ncol];
      h1s[w][row][ncol] = (_Float16)(silu_f(val) * bns + bnbv);
    }
  }
  __syncthreads();

  // ---- GEMM2: h2 = silu(h1 @ m_w2 + m_b2) ----
  for (int nt = 0; nt < 16; ++nt) {
    if (nt < 15)  // warm WGP$/L2 for the next n-tile's fragment stream
      __builtin_prefetch(w2p + ((0 * 16 + nt + 1) * 32 + lane) * 16, 0, 3);
    v8f acc = {};
#pragma unroll 4
    for (int kb = 0; kb < 16; ++kb) {
      v8h lo = *(const v8h*)&h1s[w][m][kb * 32 + g * 8];
      v8h hi = *(const v8h*)&h1s[w][m][kb * 32 + 16 + g * 8];
      v16h a;
#pragma unroll
      for (int i = 0; i < 8; ++i) { a[i] = lo[i]; a[8 + i] = hi[i]; }
      v16h b = *(const v16h*)(w2p + ((kb * 16 + nt) * 32 + lane) * 16);
      acc = __builtin_amdgcn_wmma_f32_16x16x32_f16(false, a, false, b, (short)0, acc, false, false);
    }
    int ncol = nt * 16 + m;
    float bias = m_b2[ncol];
#pragma unroll
    for (int v = 0; v < 8; ++v)
      h2s[w][v + 8 * g][ncol] = (_Float16)silu_f(acc[v] + bias);
  }
  __syncthreads();

  // ---- GEMM3: pred = h2 @ m_w3 + m_b3 ; fused squared-error vs noise ----
  float part[8];
#pragma unroll
  for (int v = 0; v < 8; ++v) part[v] = 0.f;
  for (int nt = 0; nt < 4; ++nt) {
    v8f acc = {};
#pragma unroll
    for (int kb = 0; kb < 8; ++kb) {
      v8h lo = *(const v8h*)&h2s[w][m][kb * 32 + g * 8];
      v8h hi = *(const v8h*)&h2s[w][m][kb * 32 + 16 + g * 8];
      v16h a;
#pragma unroll
      for (int i = 0; i < 8; ++i) { a[i] = lo[i]; a[8 + i] = hi[i]; }
      v16h b = *(const v16h*)(w3p + ((kb * 4 + nt) * 32 + lane) * 16);
      acc = __builtin_amdgcn_wmma_f32_16x16x32_f16(false, a, false, b, (short)0, acc, false, false);
    }
    int ncol = nt * 16 + m;
    float bias = m_b3[ncol];
#pragma unroll
    for (int v = 0; v < 8; ++v) {
      int row = v + 8 * g;
      float pred = acc[v] + bias;
      float nz = noise[(rowbase + row) * 64 + ncol];
      float d = nz - pred;
      part[v] += d * d;
    }
  }
  // reduce each row's partial across its 16-lane group; lane (m==0) writes
#pragma unroll
  for (int v = 0; v < 8; ++v) {
    float s = part[v];
    for (int off = 1; off < 16; off <<= 1) s += __shfl_xor(s, off, 32);
    if (m == 0) {
      int rr = rowbase + v + 8 * g;
      mse_out[rr] = s * (1.f / 64.f);
    }
  }
}

// ---------------- HMM forward: one wave32, lane = 4*b + j ----------------
__global__ void __launch_bounds__(32) k_hmm(
    const float* __restrict__ trans, const float* __restrict__ start,
    const float* __restrict__ mse, float* __restrict__ out) {
  int lane = threadIdx.x;
  int b = lane >> 2, j = lane & 3;
  // log_softmax columns this lane needs
  float ltj[4];
#pragma unroll
  for (int i = 0; i < 4; ++i) {
    float r0 = trans[i * 4 + 0], r1 = trans[i * 4 + 1];
    float r2 = trans[i * 4 + 2], r3 = trans[i * 4 + 3];
    float mx = fmaxf(fmaxf(r0, r1), fmaxf(r2, r3));
    float lse = mx + __logf(__expf(r0 - mx) + __expf(r1 - mx) + __expf(r2 - mx) + __expf(r3 - mx));
    ltj[i] = trans[i * 4 + j] - lse;
  }
  float s0 = start[0], s1 = start[1], s2 = start[2], s3 = start[3];
  float smx = fmaxf(fmaxf(s0, s1), fmaxf(s2, s3));
  float slse = smx + __logf(__expf(s0 - smx) + __expf(s1 - smx) + __expf(s2 - smx) + __expf(s3 - smx));
  float lsj = start[j] - slse;

  int n0 = b * Ll;
  float e0 = -(100.f / 3.f) * (mse[(j * 3 + 0) * Nn + n0] +
                               mse[(j * 3 + 1) * Nn + n0] +
                               mse[(j * 3 + 2) * Nn + n0]);
  float alpha = lsj + e0;
  out[(b * Ll + 0) * 4 + j] = alpha;
  for (int l = 1; l < Ll; ++l) {
    float a0 = __shfl(alpha, b * 4 + 0, 32);
    float a1 = __shfl(alpha, b * 4 + 1, 32);
    float a2 = __shfl(alpha, b * 4 + 2, 32);
    float a3 = __shfl(alpha, b * 4 + 3, 32);
    float v0 = a0 + ltj[0], v1 = a1 + ltj[1], v2 = a2 + ltj[2], v3 = a3 + ltj[3];
    float mx = fmaxf(fmaxf(v0, v1), fmaxf(v2, v3));
    float lse = mx + __logf(__expf(v0 - mx) + __expf(v1 - mx) + __expf(v2 - mx) + __expf(v3 - mx));
    int n = n0 + l;
    float e = -(100.f / 3.f) * (mse[(j * 3 + 0) * Nn + n] +
                                mse[(j * 3 + 1) * Nn + n] +
                                mse[(j * 3 + 2) * Nn + n]);
    alpha = lse + e;
    out[(b * Ll + l) * 4 + j] = alpha;
  }
}

// ---------------- launcher ----------------
extern "C" void kernel_launch(void* const* d_in, const int* in_sizes, int n_in,
                              void* d_out, int out_size, void* d_ws, size_t ws_size,
                              hipStream_t stream) {
  (void)in_sizes; (void)n_in; (void)out_size; (void)ws_size;
  const float* x      = (const float*)d_in[0];
  const int*   t_idx  = (const int*)d_in[1];
  const float* noise  = (const float*)d_in[2];
  const float* t_w1   = (const float*)d_in[3];
  const float* t_b1   = (const float*)d_in[4];
  const float* t_w2   = (const float*)d_in[5];
  const float* t_b2   = (const float*)d_in[6];
  const float* emb    = (const float*)d_in[7];
  const float* p_w    = (const float*)d_in[8];
  const float* p_b    = (const float*)d_in[9];
  const float* m_w1   = (const float*)d_in[10];
  const float* m_b1   = (const float*)d_in[11];
  const float* bn_g   = (const float*)d_in[12];
  const float* bn_b   = (const float*)d_in[13];
  const float* m_w2   = (const float*)d_in[14];
  const float* m_b2   = (const float*)d_in[15];
  const float* m_w3   = (const float*)d_in[16];
  const float* m_b3   = (const float*)d_in[17];
  const float* trans  = (const float*)d_in[18];
  const float* start  = (const float*)d_in[19];

  float* ws = (float*)d_ws;
  _Float16* w1p = (_Float16*)((char*)d_ws + W1P_B);
  _Float16* w2p = (_Float16*)((char*)d_ws + W2P_B);
  _Float16* w3p = (_Float16*)((char*)d_ws + W3P_B);

  k_sched  <<<1,   128, 0, stream>>>(bn_g, ws);
  k_c0kt   <<<5,   512, 0, stream>>>(p_b, m_b1, emb, m_w1, ws);
  k_ttable <<<100, 512, 0, stream>>>(t_w1, t_b1, t_w2, t_b2, m_w1, ws);
  k_w1pack <<<64,  512, 0, stream>>>(p_w, m_w1, w1p);
  k_w2pack <<<512, 256, 0, stream>>>(m_w2, w2p);
  k_w3pack <<<256, 64,  0, stream>>>(m_w3, w3p);
  k_main   <<<KSN / 32, 64, 0, stream>>>(x, t_idx, noise, bn_b, m_b2, m_b3,
                                         ws, w1p, w2p, w3p, ws + MSEO);
  k_hmm    <<<1, 32, 0, stream>>>(trans, start, ws + MSEO, (float*)d_out);
}
